// LonelyDecoder_33964601376784
// MI455X (gfx1250) — matrix operand order repaired and checked by reading, not compiled, not run on hardware
//
#include <hip/hip_runtime.h>
#include <hip/hip_bf16.h>

// ---------------------------------------------------------------------------
// Problem constants (from reference)
// ---------------------------------------------------------------------------
#define S_   2048
#define V_   32000
#define D_   1024
#define H_   16
#define DK_  64
#define DFF_ 4096
#define L_   4

// GEMM tiling
#define BM 128
#define BN 128
#define BK 32
#define BKP 40   // padded LDS row (bank-conflict avoidance, keeps 16B alignment)

typedef __attribute__((ext_vector_type(16))) _Float16 v16h;
typedef __attribute__((ext_vector_type(8)))  _Float16 v8h;
typedef __attribute__((ext_vector_type(8)))  float    v8f;

enum { GF_RELU = 1, GF_CAUSAL = 2, GF_OUTF16 = 4, GF_TRANSB = 8, GF_BIAS = 16 };

// ---------------------------------------------------------------------------
// CDNA5 async global->LDS path (guarded: falls back to sync staging if the
// toolchain lacks the builtin).
// ---------------------------------------------------------------------------
#if defined(__gfx1250__) && __has_builtin(__builtin_amdgcn_global_load_async_to_lds_b128)
#define HAVE_ASYNC_LDS 1
#else
#define HAVE_ASYNC_LDS 0
#endif

#if HAVE_ASYNC_LDS
// Builtin (per hipcc diagnostic) takes: (int4 AS1* gsrc, int4 AS3* ldsdst, imm, imm)
typedef int v4i_ __attribute__((__vector_size__(4 * sizeof(int))));
typedef __attribute__((address_space(1))) v4i_ g_v4i;
typedef __attribute__((address_space(3))) v4i_ l_v4i;

__device__ __forceinline__ void async_b128(void* lds, const void* g) {
  __builtin_amdgcn_global_load_async_to_lds_b128(
      (g_v4i*)(unsigned long long)(__SIZE_TYPE__)g,
      (l_v4i*)(unsigned int)(__SIZE_TYPE__)lds, 0, 0);
}

__device__ __forceinline__ void wait_async0() {
#if __has_builtin(__builtin_amdgcn_s_wait_asynccnt)
  __builtin_amdgcn_s_wait_asynccnt(0);
#else
  asm volatile("s_wait_asynccnt 0" ::: "memory");
#endif
}
#endif

// ---------------------------------------------------------------------------
// f32 -> f16 conversion (grid-stride)
// ---------------------------------------------------------------------------
__global__ __launch_bounds__(256) void f32_to_f16_k(const float* __restrict__ in,
                                                    _Float16* __restrict__ out,
                                                    long n) {
  for (long i = (long)blockIdx.x * 256 + threadIdx.x; i < n; i += (long)gridDim.x * 256)
    out[i] = (_Float16)in[i];
}

// ---------------------------------------------------------------------------
// Generic batched f16 WMMA GEMM: C = act(scale * A@B(^T) + bias) [+ causal mask]
//   A: [M,K] f16, lda;  B: [K,N] f16 (or [N,K] if GF_TRANSB), ldb
//   C: f32 or f16, ldc. M multiple of 128, K multiple of 32 (true here).
//   N any multiple of 8 (guarded). Double-buffered LDS, async staging.
// ---------------------------------------------------------------------------
__global__ __launch_bounds__(256) void gemm_f16_wmma(
    const _Float16* __restrict__ A, long sAb, int lda,
    const _Float16* __restrict__ B, long sBb, int ldb,
    const float* __restrict__ bias, long sBias,
    void* __restrict__ Cv, long sCb, int ldc,
    int N, int K, float scale, int flags) {
  __shared__ _Float16 As[2][BM][BKP];
  __shared__ _Float16 Bs[2][BN][BKP];

  const int tid = threadIdx.x;
  const int z   = blockIdx.z;
  A += sAb * z;
  B += sBb * z;
  float*    Cf = (float*)Cv + sCb * z;
  _Float16* Ch = (_Float16*)Cv + sCb * z;

  const int rowBase = blockIdx.y * BM;
  const int colBase = blockIdx.x * BN;

  const int wid   = tid >> 5;
  const int lane  = tid & 31;
  const int wm    = wid & 3;    // 0..3 : 32-row strip
  const int wn    = wid >> 2;   // 0..1 : 64-col strip
  const int lhalf = lane >> 4;  // 0/1
  const int l16   = lane & 15;

  // ---- tile staging (one K-slab into LDS buffer `buf`) ----
  auto stageA = [&](int k0, int buf) {
    const int r  = tid >> 1;
    const int kc = (tid & 1) * 16;
    const _Float16* src = A + (long)(rowBase + r) * lda + k0 + kc;
#if HAVE_ASYNC_LDS
    async_b128(&As[buf][r][kc], src);
    async_b128(&As[buf][r][kc + 8], src + 8);
#else
    v8h a0 = *(const v8h*)src;
    v8h a1 = *(const v8h*)(src + 8);
    *(v8h*)&As[buf][r][kc]     = a0;
    *(v8h*)&As[buf][r][kc + 8] = a1;
#endif
    if (k0 + 2 * BK < K) __builtin_prefetch(src + 2 * BK, 0, 1);
  };
  auto stageB = [&](int k0, int buf) {
    if (flags & GF_TRANSB) {
      const int n  = tid >> 1;
      const int kc = (tid & 1) * 16;
      if (colBase + n < N) {
        const _Float16* src = B + (long)(colBase + n) * ldb + k0 + kc;
#if HAVE_ASYNC_LDS
        async_b128(&Bs[buf][n][kc], src);
        async_b128(&Bs[buf][n][kc + 8], src + 8);
#else
        v8h b0 = *(const v8h*)src;
        v8h b1 = *(const v8h*)(src + 8);
        *(v8h*)&Bs[buf][n][kc]     = b0;
        *(v8h*)&Bs[buf][n][kc + 8] = b1;
#endif
        if (k0 + 2 * BK < K) __builtin_prefetch(src + 2 * BK, 0, 1);
      } else {
        v8h zv = {};
        *(v8h*)&Bs[buf][n][kc]     = zv;
        *(v8h*)&Bs[buf][n][kc + 8] = zv;
      }
    } else {
      // B [K,N]: transpose into K-contiguous-per-column LDS layout
      const int kBaseT = tid >> 4;        // 0..15
      const int n8     = (tid & 15) * 8;  // 0..120
#pragma unroll
      for (int kk = kBaseT; kk < BK; kk += 16) {
        v8h b = {};
        if (colBase + n8 < N) {
          const _Float16* src = B + (long)(k0 + kk) * ldb + colBase + n8;
          b = *(const v8h*)src;
          if (k0 + 2 * BK < K) __builtin_prefetch(src + (long)2 * BK * ldb, 0, 1);
        }
#pragma unroll
        for (int i = 0; i < 8; ++i) Bs[buf][n8 + i][kk] = b[i];
      }
    }
  };

  v8f acc[2][4];
  v8f zero = {};
#pragma unroll
  for (int tm = 0; tm < 2; ++tm)
#pragma unroll
    for (int tn = 0; tn < 4; ++tn) acc[tm][tn] = zero;

  const int nk = K / BK;
  stageA(0, 0);
  stageB(0, 0);

  for (int kt = 0; kt < nk; ++kt) {
    const int cur = kt & 1;
#if HAVE_ASYNC_LDS
    wait_async0();            // own async writes into buf[cur] done
#endif
    __syncthreads();          // all waves' staging of buf[cur] visible

    if (kt + 1 < nk) {        // overlap: stage next slab into other buffer
      stageA((kt + 1) * BK, cur ^ 1);
      stageB((kt + 1) * BK, cur ^ 1);
    }

    // ---- fragment loads per ISA 16-bit layouts ----
    v16h af[2], bf[4];
#pragma unroll
    for (int tm = 0; tm < 2; ++tm) {
      const int m = wm * 32 + tm * 16 + l16;
      v8h lo = *(const v8h*)&As[cur][m][lhalf * 8];
      v8h hi = *(const v8h*)&As[cur][m][16 + lhalf * 8];
#pragma unroll
      for (int i = 0; i < 8; ++i) { af[tm][i] = lo[i]; af[tm][i + 8] = hi[i]; }
    }
#pragma unroll
    for (int tn = 0; tn < 4; ++tn) {
      const int n = wn * 64 + tn * 16 + l16;
      v8h lo = *(const v8h*)&Bs[cur][n][lhalf * 8];
      v8h hi = *(const v8h*)&Bs[cur][n][16 + lhalf * 8];
#pragma unroll
      for (int i = 0; i < 8; ++i) { bf[tn][i] = lo[i]; bf[tn][i + 8] = hi[i]; }
    }
    // ---- 2x4 WMMA ----
#pragma unroll
    for (int tm = 0; tm < 2; ++tm)
#pragma unroll
      for (int tn = 0; tn < 4; ++tn)
        acc[tm][tn] = __builtin_amdgcn_wmma_f32_16x16x32_f16(
            false, af[tm], false, bf[tn], (short)0, acc[tm][tn], false, false);
  }

  // ---- epilogue: scale, bias, mask, activation, store ----
#pragma unroll
  for (int tm = 0; tm < 2; ++tm) {
#pragma unroll
    for (int tn = 0; tn < 4; ++tn) {
      const int col = colBase + wn * 64 + tn * 16 + l16;
      if (col >= N) continue;
      float bv = 0.f;
      if (flags & GF_BIAS) bv = bias[sBias * z + col];
#pragma unroll
      for (int j = 0; j < 8; ++j) {
        const int row = rowBase + wm * 32 + tm * 16 + j + 8 * lhalf;
        float val = acc[tm][tn][j] * scale + bv;
        if ((flags & GF_CAUSAL) && col > row) val = -1e9f;
        if (flags & GF_RELU) val = fmaxf(val, 0.f);
        const long idx = (long)row * ldc + col;
        if (flags & GF_OUTF16) Ch[idx] = (_Float16)val;
        else                   Cf[idx] = val;
      }
    }
  }
}

// ---------------------------------------------------------------------------
// Row softmax (one block per row). outF16: write f16, else f32 (may be inplace)
// ---------------------------------------------------------------------------
__global__ __launch_bounds__(256) void softmax_rows_k(const float* __restrict__ in,
                                                      void* __restrict__ out,
                                                      int cols, int outF16) {
  __shared__ float red[256];
  const int row = blockIdx.x;
  const float* x = in + (long)row * cols;

  float m = -3.4e38f;
  for (int c = threadIdx.x; c < cols; c += 256) m = fmaxf(m, x[c]);
  red[threadIdx.x] = m;
  __syncthreads();
  for (int s = 128; s > 0; s >>= 1) {
    if (threadIdx.x < s) red[threadIdx.x] = fmaxf(red[threadIdx.x], red[threadIdx.x + s]);
    __syncthreads();
  }
  m = red[0];
  __syncthreads();

  float sum = 0.f;
  for (int c = threadIdx.x; c < cols; c += 256) sum += __expf(x[c] - m);
  red[threadIdx.x] = sum;
  __syncthreads();
  for (int s = 128; s > 0; s >>= 1) {
    if (threadIdx.x < s) red[threadIdx.x] += red[threadIdx.x + s];
    __syncthreads();
  }
  const float inv = 1.f / red[0];

  if (outF16) {
    _Float16* o = (_Float16*)out + (long)row * cols;
    for (int c = threadIdx.x; c < cols; c += 256) o[c] = (_Float16)(__expf(x[c] - m) * inv);
  } else {
    float* o = (float*)out + (long)row * cols;
    for (int c = threadIdx.x; c < cols; c += 256) o[c] = __expf(x[c] - m) * inv;
  }
}

// ---------------------------------------------------------------------------
// y = LayerNorm(a + b) * g + beta ; writes f32 and f16 mirror. One block/row.
// ---------------------------------------------------------------------------
__global__ __launch_bounds__(256) void add_layernorm_k(
    const float* __restrict__ A, const float* __restrict__ Bv,
    const float* __restrict__ g, const float* __restrict__ be,
    float* __restrict__ Y, _Float16* __restrict__ Yh) {
  __shared__ float red[256];
  const int row = blockIdx.x;
  const float* a = A + (long)row * D_;
  const float* b = Bv + (long)row * D_;

  float v[4], s = 0.f;
#pragma unroll
  for (int j = 0; j < 4; ++j) {
    const int c = threadIdx.x + j * 256;
    v[j] = a[c] + b[c];
    s += v[j];
  }
  red[threadIdx.x] = s;
  __syncthreads();
  for (int t = 128; t > 0; t >>= 1) {
    if (threadIdx.x < t) red[threadIdx.x] += red[threadIdx.x + t];
    __syncthreads();
  }
  const float mu = red[0] * (1.f / D_);
  __syncthreads();

  float vs = 0.f;
#pragma unroll
  for (int j = 0; j < 4; ++j) { const float d = v[j] - mu; vs += d * d; }
  red[threadIdx.x] = vs;
  __syncthreads();
  for (int t = 128; t > 0; t >>= 1) {
    if (threadIdx.x < t) red[threadIdx.x] += red[threadIdx.x + t];
    __syncthreads();
  }
  const float rinv = rsqrtf(red[0] * (1.f / D_) + 1e-5f);

#pragma unroll
  for (int j = 0; j < 4; ++j) {
    const int c = threadIdx.x + j * 256;
    const float y = (v[j] - mu) * rinv * g[c] + be[c];
    Y[(long)row * D_ + c]  = y;
    Yh[(long)row * D_ + c] = (_Float16)y;
  }
}

// ---------------------------------------------------------------------------
// h += emb_b + positional encoding ; write f16 mirror
// ---------------------------------------------------------------------------
__global__ __launch_bounds__(256) void add_bias_pe_k(float* __restrict__ h,
                                                     _Float16* __restrict__ hh,
                                                     const float* __restrict__ emb_b) {
  const long n = (long)S_ * D_;
  for (long i = (long)blockIdx.x * 256 + threadIdx.x; i < n; i += (long)gridDim.x * 256) {
    const int s = (int)(i / D_);
    const int d = (int)(i % D_);
    const int i2 = d & ~1;
    const float ang = (float)s * __expf(-9.210340371976184f * (float)i2 / (float)D_);
    const float pe = (d & 1) ? __cosf(ang) : __sinf(ang);
    const float v = h[i] + emb_b[d] + pe;
    h[i] = v;
    hh[i] = (_Float16)v;
  }
}

// ---------------------------------------------------------------------------
// Host orchestration
// ---------------------------------------------------------------------------
extern "C" void kernel_launch(void* const* d_in, const int* in_sizes, int n_in,
                              void* d_out, int out_size, void* d_ws, size_t ws_size,
                              hipStream_t stream) {
  (void)in_sizes; (void)n_in; (void)out_size; (void)ws_size;

  const float* x     = (const float*)d_in[0];
  const float* emb_W = (const float*)d_in[1];
  const float* emb_b = (const float*)d_in[2];
  const float* Wq1   = (const float*)d_in[3];
  const float* bq1   = (const float*)d_in[4];
  const float* Wk1   = (const float*)d_in[5];
  const float* bk1   = (const float*)d_in[6];
  const float* Wv1   = (const float*)d_in[7];
  const float* bv1   = (const float*)d_in[8];
  const float* Wq2   = (const float*)d_in[9];
  const float* bq2   = (const float*)d_in[10];
  const float* Wk2   = (const float*)d_in[11];
  const float* bk2   = (const float*)d_in[12];
  const float* Wv2   = (const float*)d_in[13];
  const float* bv2   = (const float*)d_in[14];
  const float* ffW1  = (const float*)d_in[15];
  const float* ffb1  = (const float*)d_in[16];
  const float* ffW2  = (const float*)d_in[17];
  const float* ffb2  = (const float*)d_in[18];
  const float* ln_g  = (const float*)d_in[19];
  const float* ln_b  = (const float*)d_in[20];
  const float* out_W = (const float*)d_in[21];
  const float* out_b = (const float*)d_in[22];

  // ---- workspace layout ----
  char* p = (char*)d_ws;
  auto alloc = [&](size_t bytes) -> void* {
    void* r = (void*)p;
    p += (bytes + 255) & ~(size_t)255;
    return r;
  };
  _Float16* xb   = (_Float16*)alloc((size_t)S_ * V_ * 2);   // x in f16
  _Float16* wbuf = (_Float16*)alloc((size_t)V_ * D_ * 2);   // shared weight f16 buffer
  float*    h    = (float*)   alloc((size_t)S_ * D_ * 4);
  _Float16* hh   = (_Float16*)alloc((size_t)S_ * D_ * 2);
  float*    h1   = (float*)   alloc((size_t)S_ * D_ * 4);
  _Float16* h1h  = (_Float16*)alloc((size_t)S_ * D_ * 2);
  float*    h2   = (float*)   alloc((size_t)S_ * D_ * 4);
  _Float16* h2h  = (_Float16*)alloc((size_t)S_ * D_ * 2);
  float*    ofin = (float*)   alloc((size_t)S_ * D_ * 4);   // final LN out (f32)
  _Float16* ofih = (_Float16*)alloc((size_t)S_ * D_ * 2);
  float*    tmp  = (float*)   alloc((size_t)S_ * D_ * 4);   // mha / ffn2 output
  _Float16* q    = (_Float16*)alloc((size_t)S_ * D_ * 2);   // [H][S][DK]
  _Float16* kk_  = (_Float16*)alloc((size_t)S_ * D_ * 2);
  _Float16* vv   = (_Float16*)alloc((size_t)S_ * D_ * 2);
  _Float16* ffh  = (_Float16*)alloc((size_t)S_ * DFF_ * 2); // relu(ffn1) in f16
  float*    sc   = (float*)   alloc((size_t)S_ * S_ * 4);   // per-head scores
  _Float16* pr   = (_Float16*)alloc((size_t)S_ * S_ * 2);   // per-head probs

  auto cvt = [&](const float* src, _Float16* dst, long n) {
    f32_to_f16_k<<<4096, 256, 0, stream>>>(src, dst, n);
  };
  auto gemm = [&](const _Float16* A, long sAb, int lda,
                  const _Float16* B, long sBb, int ldb,
                  const float* bias, long sBias,
                  void* C, long sCb, int ldc,
                  int M, int N, int K, float scale, int flags, int batch) {
    dim3 grid((N + BN - 1) / BN, M / BM, batch);
    gemm_f16_wmma<<<grid, 256, 0, stream>>>(A, sAb, lda, B, sBb, ldb, bias, sBias,
                                            C, sCb, ldc, N, K, scale, flags);
  };

  // ---- embedding: h = x @ emb_W, then + emb_b + PE ----
  cvt(x, xb, (long)S_ * V_);
  cvt(emb_W, wbuf, (long)V_ * D_);
  gemm(xb, 0, V_, wbuf, 0, D_, nullptr, 0, h, 0, D_, S_, D_, V_, 1.f, 0, 1);
  add_bias_pe_k<<<4096, 256, 0, stream>>>(h, hh, emb_b);

  // ---- only the LAST layer's output feeds the head (loop carries no state) ----
  const int li = L_ - 1;
  const long wOffQKV = (long)li * H_ * D_ * DK_;
  const long bOffQKV = (long)li * H_ * DK_;

  auto mha = [&](const _Float16* xh, const float* Wq, const float* bq,
                 const float* Wk, const float* bk, const float* Wv, const float* bv,
                 bool masked, float* outp) {
    // Q/K/V projections, batched over heads (N = DK = 64, guarded)
    cvt(Wq, wbuf, (long)H_ * D_ * DK_);
    gemm(xh, 0, D_, wbuf, (long)D_ * DK_, DK_, bq, DK_, q, (long)S_ * DK_, DK_,
         S_, DK_, D_, 1.f, GF_BIAS | GF_OUTF16, H_);
    cvt(Wk, wbuf, (long)H_ * D_ * DK_);
    gemm(xh, 0, D_, wbuf, (long)D_ * DK_, DK_, bk, DK_, kk_, (long)S_ * DK_, DK_,
         S_, DK_, D_, 1.f, GF_BIAS | GF_OUTF16, H_);
    cvt(Wv, wbuf, (long)H_ * D_ * DK_);
    gemm(xh, 0, D_, wbuf, (long)D_ * DK_, DK_, bv, DK_, vv, (long)S_ * DK_, DK_,
         S_, DK_, D_, 1.f, GF_BIAS | GF_OUTF16, H_);

    for (int hd = 0; hd < H_; ++hd) {
      const _Float16* qh = q   + (long)hd * S_ * DK_;
      const _Float16* kh = kk_ + (long)hd * S_ * DK_;
      const _Float16* vh = vv  + (long)hd * S_ * DK_;
      // scores = q @ k^T / D (+ causal mask); note: reference scales by 1/D
      gemm(qh, 0, DK_, kh, 0, DK_, nullptr, 0, sc, 0, S_,
           S_, S_, DK_, 1.f / (float)D_, GF_TRANSB | (masked ? GF_CAUSAL : 0), 1);
      softmax_rows_k<<<S_, 256, 0, stream>>>(sc, pr, S_, 1);
      // o_head = probs @ v  (written into concat layout at column hd*DK)
      gemm(pr, 0, S_, vh, 0, DK_, nullptr, 0, outp + hd * DK_, 0, D_,
           S_, DK_, S_, 1.f, 0, 1);
    }
  };

  // h1 = LN(h + mha_masked(h))
  mha(hh, Wq1 + wOffQKV, bq1 + bOffQKV, Wk1 + wOffQKV, bk1 + bOffQKV,
      Wv1 + wOffQKV, bv1 + bOffQKV, true, tmp);
  add_layernorm_k<<<S_, 256, 0, stream>>>(h, tmp, ln_g, ln_b, h1, h1h);

  // h2 = LN(h1 + mha(h1))
  mha(h1h, Wq2 + wOffQKV, bq2 + bOffQKV, Wk2 + wOffQKV, bk2 + bOffQKV,
      Wv2 + wOffQKV, bv2 + bOffQKV, false, tmp);
  add_layernorm_k<<<S_, 256, 0, stream>>>(h1, tmp, ln_g, ln_b, h2, h2h);

  // ffn: relu(h2 @ W1 + b1) @ W2 + b2 ; out = LN(h2 + ffn)
  cvt(ffW1 + (long)li * D_ * DFF_, wbuf, (long)D_ * DFF_);
  gemm(h2h, 0, D_, wbuf, 0, DFF_, ffb1 + (long)li * DFF_, 0, ffh, 0, DFF_,
       S_, DFF_, D_, 1.f, GF_BIAS | GF_RELU | GF_OUTF16, 1);
  cvt(ffW2 + (long)li * DFF_ * D_, wbuf, (long)DFF_ * D_);
  gemm(ffh, 0, DFF_, wbuf, 0, D_, ffb2 + (long)li * D_, 0, tmp, 0, D_,
       S_, D_, DFF_, 1.f, GF_BIAS, 1);
  add_layernorm_k<<<S_, 256, 0, stream>>>(h2, tmp, ln_g, ln_b, ofin, ofih);

  // ---- logits = out @ out_W + out_b ; softmax over V in place in d_out ----
  cvt(out_W, wbuf, (long)D_ * V_);
  gemm(ofih, 0, D_, wbuf, 0, V_, out_b, 0, d_out, 0, V_,
       S_, V_, D_, 1.f, GF_BIAS, 1);
  softmax_rows_k<<<S_, 256, 0, stream>>>((const float*)d_out, d_out, V_, 0);
}
